// GridNetBlock_43739946943058
// MI455X (gfx1250) — matrix operand order repaired and checked by reading, not compiled, not run on hardware
//
#include <hip/hip_runtime.h>
#include <hip/hip_bf16.h>

typedef __attribute__((ext_vector_type(16))) _Float16 v16h;
typedef __attribute__((ext_vector_type(8)))  _Float16 v8h;
typedef __attribute__((ext_vector_type(8)))  float    v8f;

#define Bn   2
#define Cc   48
#define Tt   512
#define Ffq  65
#define Hh   192
#define CIN  192     // C*K (unfold kernel size 4)
#define G4   768     // 4*H
#define LA   62      // intra unfold length (F dim)
#define NA   1024    // B*T
#define LB   509     // inter unfold length (T dim)
#define NB   130     // B*F
#define NBP  144     // padded to multiple of 16
#define NHh  4
#define Ee   8
#define Dd   12
#define EF   520     // E*F
#define EFP  544     // padded K for WMMA (mult of 32)
#define DF   780     // D*F
#define DFP  784     // padded N (mult of 16)
#define EPSV 1e-5f

// dynamic LDS for the recurrence kernel: Wh(288KB) + h(6KB) + gates f16(24KB)
#define LSTM_LDS_WH   ((size_t)G4 * Hh * 2)          // 294912
#define LSTM_LDS_H    ((size_t)16 * Hh * 2)          // 6144
#define LSTM_LDS_G    ((size_t)16 * G4 * 2)          // 24576
#define LSTM_LDS_TOT  (LSTM_LDS_WH + LSTM_LDS_H + LSTM_LDS_G)  // 325632 <= 320KB

// ---------------------------------------------------------------------------
// WMMA helpers (gfx1250 wave32, 16x16x32 f16 -> f32)
// ---------------------------------------------------------------------------
__device__ __forceinline__ v8f wmma16(v16h a, v16h b, v8f c) {
  return __builtin_amdgcn_wmma_f32_16x16x32_f16(false, a, false, b, (short)0, c,
                                                false, false);
}

// A-matrix fragment (16x32 f16, row-major source, ldk halfs per row).
// Lane m=lane%16 holds row m; halves cover K = {hi*8..+7} U {16+hi*8..+7}
__device__ __forceinline__ v16h frag_a(const _Float16* base, int ldk, int m,
                                       int hi, int kt) {
  const _Float16* p = base + (size_t)m * ldk + kt * 32 + hi * 8;
  v8h lo = *(const v8h*)(p);
  v8h h8 = *(const v8h*)(p + 16);
  v16h r;
#pragma unroll
  for (int i = 0; i < 8; ++i) { r[i] = lo[i]; r[i + 8] = h8[i]; }
  return r;
}

// B-matrix fragment (32x16 f16). Source holds B TRANSPOSED row-major:
// row n (output column), contiguous along K. K = {hi*8..+7} U {16+hi*8..+7}.
__device__ __forceinline__ v16h frag_b(const _Float16* base, int ldk, int n,
                                       int hi, int kt) {
  const _Float16* p = base + (size_t)n * ldk + kt * 32 + hi * 8;
  v8h lo = *(const v8h*)(p);
  v8h h8 = *(const v8h*)(p + 16);
  v16h r;
#pragma unroll
  for (int i = 0; i < 8; ++i) { r[i] = lo[i]; r[i + 8] = h8[i]; }
  return r;
}

__device__ __forceinline__ float sigf(float x) { return 1.0f / (1.0f + __expf(-x)); }

__device__ __forceinline__ float blk_sum(float v, float* red) {
  int tid = threadIdx.x;
  red[tid] = v;
  __syncthreads();
  for (int s = 128; s > 0; s >>= 1) {
    if (tid < s) red[tid] += red[tid + s];
    __syncthreads();
  }
  float r = red[0];
  __syncthreads();
  return r;
}

// ---------------------------------------------------------------------------
// Weight preparation (recomputed every call - deterministic, no caching)
// ---------------------------------------------------------------------------
__global__ void k_f2h(const float* __restrict__ s, _Float16* __restrict__ d, int n) {
  int i = blockIdx.x * 256 + threadIdx.x;
  if (i < n) d[i] = (_Float16)s[i];
}

// lw (2H,C,K)=(384,48,4) f32 -> lwT[c][k][j] f16 (contiguous over j)
__global__ void k_lwT(const float* __restrict__ lw, _Float16* __restrict__ lwT) {
  int i = blockIdx.x * 256 + threadIdx.x;
  if (i >= 384 * 48 * 4) return;
  int j = i / (48 * 4), c = (i / 4) % 48, k = i % 4;
  lwT[(((size_t)c) * 4 + k) * 384 + j] = (_Float16)lw[i];
}

__global__ void k_bsum(const float* __restrict__ a, const float* __restrict__ b,
                       float* __restrict__ o, int n) {
  int i = blockIdx.x * 256 + threadIdx.x;
  if (i < n) o[i] = a[i] + b[i];
}

// ---------------------------------------------------------------------------
// LayerNorm over C (per b,t,f) + unfold along F -> xs f16 [n=bT+t][l][192]
// ---------------------------------------------------------------------------
__global__ __launch_bounds__(256) void k_ln_unfold_intra(
    const float* __restrict__ x, const float* __restrict__ gam,
    const float* __restrict__ bet, _Float16* __restrict__ xs) {
  __shared__ float sx[Cc][Ffq];
  __shared__ float hn[Cc][Ffq];
  __shared__ float smu[Ffq], srs[Ffq];
  const int blk = blockIdx.x;  // b*512 + t
  const int b = blk >> 9, t = blk & 511;
  for (int i = threadIdx.x; i < Cc * Ffq; i += 256) {
    int c = i / Ffq, f = i % Ffq;
    sx[c][f] = x[(((size_t)b * Cc + c) * Tt + t) * Ffq + f];
  }
  __syncthreads();
  if (threadIdx.x < Ffq) {
    int f = threadIdx.x;
    float s = 0.f, ss = 0.f;
    for (int c = 0; c < Cc; ++c) { float v = sx[c][f]; s += v; ss += v * v; }
    float mu = s / (float)Cc;
    smu[f] = mu;
    srs[f] = rsqrtf(ss / (float)Cc - mu * mu + EPSV);
  }
  __syncthreads();
  for (int i = threadIdx.x; i < Cc * Ffq; i += 256) {
    int c = i / Ffq, f = i % Ffq;
    hn[c][f] = (sx[c][f] - smu[f]) * srs[f] * gam[c] + bet[c];
  }
  __syncthreads();
  const size_t base = (size_t)blk * LA * CIN;
  for (int i = threadIdx.x; i < LA * CIN; i += 256) {
    int l = i / CIN, ck = i % CIN, c = ck >> 2, k = ck & 3;
    xs[base + i] = (_Float16)hn[c][l + k];
  }
}

// LayerNorm over C of `intra` (per b,t,q) + unfold along T -> xs_inter
// n = b*65 + q ; rows n in [130,144) zero-filled (GEMM padding)
__global__ __launch_bounds__(256) void k_ln_unfold_inter(
    const float* __restrict__ intra, const float* __restrict__ gam,
    const float* __restrict__ bet, _Float16* __restrict__ xs) {
  const int n = blockIdx.x;
  const size_t base = (size_t)n * LB * CIN;
  if (n >= NB) {
    for (int i = threadIdx.x; i < LB * CIN; i += 256) xs[base + i] = (_Float16)0.f;
    return;
  }
  const int b = n / Ffq, qf = n % Ffq;
  __shared__ float hn[131][Cc];
  for (int ch = 0; ch < 4; ++ch) {
    const int t0 = ch * 128;
    const int nt = (Tt - t0 < 131) ? (Tt - t0) : 131;
    __syncthreads();
    if (threadIdx.x < (unsigned)nt) {
      int t = t0 + threadIdx.x;
      float vals[Cc];
      float s = 0.f, ss = 0.f;
      for (int c = 0; c < Cc; ++c) {
        float v = intra[(((size_t)b * Cc + c) * Tt + t) * Ffq + qf];
        vals[c] = v; s += v; ss += v * v;
      }
      float mu = s / (float)Cc;
      float rs = rsqrtf(ss / (float)Cc - mu * mu + EPSV);
      for (int c = 0; c < Cc; ++c) hn[threadIdx.x][c] = (vals[c] - mu) * rs * gam[c] + bet[c];
    }
    __syncthreads();
    const int nl = (LB - t0 < 128) ? (LB - t0) : 128;
    for (int i = threadIdx.x; i < nl * CIN; i += 256) {
      int li = i / CIN, ck = i % CIN, c = ck >> 2, k = ck & 3;
      xs[base + (size_t)(t0 + li) * CIN + ck] = (_Float16)hn[li + k][c];
    }
  }
}

// ---------------------------------------------------------------------------
// Input projection GEMM (no sequential dependency):
// xg[dir][row][g] = xs[row]@Wi[dir]^T + (bi+bh)[dir][g], rows = N*L flat.
// grid (Mrows/16, 6, 2); 8 waves/block, one 16-col tile per wave.
// ---------------------------------------------------------------------------
__global__ __launch_bounds__(256) void k_xgemm(
    const _Float16* __restrict__ xs,     // [Mrows][192]
    const _Float16* __restrict__ wiAll,  // [2][768][192]
    const float* __restrict__ bsumAll,   // [2][768]
    _Float16* __restrict__ xg,           // [2][Mrows][768]
    int Mrows) {
  const int lane = threadIdx.x & 31, wv = threadIdx.x >> 5;
  const int m = lane & 15, hi = lane >> 4;
  const int mt = blockIdx.x;
  const int dir = blockIdx.z;
  const int ncol = (blockIdx.y * 8 + wv) * 16 + m;
  const _Float16* wi = wiAll + (size_t)dir * G4 * CIN;
  const _Float16* ap = xs + (size_t)mt * 16 * CIN;
  v8f acc = {};
#pragma unroll
  for (int kt = 0; kt < 6; ++kt)
    acc = wmma16(frag_a(ap, CIN, m, hi, kt), frag_b(wi, CIN, ncol, hi, kt), acc);
  const float bias = bsumAll[dir * G4 + ncol];
  _Float16* op = xg + ((size_t)dir * Mrows + (size_t)mt * 16) * G4;
#pragma unroll
  for (int r = 0; r < 8; ++r)
    op[(size_t)(r + 8 * hi) * G4 + ncol] = (_Float16)(acc[r] + bias);
}

// ---------------------------------------------------------------------------
// LSTM recurrence only: g_t = xg_t + h_{t-1}@Wh^T, gate nonlinearity.
// One block = 16 sequences; Wh lives in LDS (288KB of the 320KB/WGP budget),
// h-state in LDS (A-fragment source), c-state in registers.
// Dynamic LDS layout: [Wh f16 768x192][h f16 16x192][gates f16 16x768]
// grid = (N/16, 2 directions)
// ---------------------------------------------------------------------------
__global__ __launch_bounds__(256) void k_lstm_rec(
    const _Float16* __restrict__ xg,     // [2][N*L][768]
    const _Float16* __restrict__ whAll,  // [2][768][192]
    _Float16* __restrict__ hs,           // [N][L][384] (fwd 0:192, bwd 192:384)
    int N, int L) {
  extern __shared__ char smem[];
  _Float16* sWh = (_Float16*)smem;
  _Float16* sH  = (_Float16*)(smem + LSTM_LDS_WH);
  _Float16* sG  = (_Float16*)(smem + LSTM_LDS_WH + LSTM_LDS_H);
  const int tid = threadIdx.x;
  const int lane = tid & 31, wv = tid >> 5;
  const int m = lane & 15, hi = lane >> 4;
  const int dir = blockIdx.y;
  const int nbase = blockIdx.x * 16;
  const int urow = tid >> 4, uj0 = tid & 15;  // elementwise phase mapping
  const _Float16* xgd = xg + (size_t)dir * ((size_t)N * L) * G4;
  // stage Wh[dir] into LDS (one-time; 73728 dwords)
  {
    const unsigned int* src = (const unsigned int*)(whAll + (size_t)dir * G4 * Hh);
    unsigned int* dst = (unsigned int*)sWh;
    for (int i = tid; i < G4 * Hh / 2; i += 256) dst[i] = src[i];
  }
  for (int i = tid; i < 16 * Hh; i += 256) sH[i] = (_Float16)0.f;
  float creg[12];
#pragma unroll
  for (int i = 0; i < 12; ++i) creg[i] = 0.f;
  __syncthreads();
  for (int s = 0; s < L; ++s) {
    const int t = dir ? (L - 1 - s) : s;
    // GEMM phase: each wave computes 6 column tiles (96 gate columns)
#pragma clang loop unroll(disable)
    for (int c6 = 0; c6 < 6; ++c6) {
      const int ncol = (wv * 6 + c6) * 16 + m;
      v8f acc = {};
#pragma unroll
      for (int kt = 0; kt < 6; ++kt)
        acc = wmma16(frag_a(sH, Hh, m, hi, kt), frag_b(sWh, Hh, ncol, hi, kt), acc);
#pragma unroll
      for (int r = 0; r < 8; ++r) {
        float v = acc[r] +
                  (float)xgd[((size_t)(nbase + r + 8 * hi) * L + t) * G4 + ncol];
        sG[(r + 8 * hi) * G4 + ncol] = (_Float16)v;
      }
    }
    __syncthreads();
    // gate nonlinearity + state update; each thread owns (row, 12 j's)
#pragma unroll
    for (int ii = 0; ii < 12; ++ii) {
      const int j = uj0 + ii * 16;
      float gi = (float)sG[urow * G4 + j];
      float gf = (float)sG[urow * G4 + Hh + j];
      float gg = (float)sG[urow * G4 + 2 * Hh + j];
      float go = (float)sG[urow * G4 + 3 * Hh + j];
      float cN = sigf(gf) * creg[ii] + sigf(gi) * tanhf(gg);
      creg[ii] = cN;
      float hN = sigf(go) * tanhf(cN);
      sH[urow * Hh + j] = (_Float16)hN;
      hs[(((size_t)(nbase + urow)) * L + t) * (2 * Hh) + dir * Hh + j] = (_Float16)hN;
    }
    __syncthreads();
  }
}

// ---------------------------------------------------------------------------
// ConvTranspose1d (K=4, 2H->C) + residual.
// mode 0 (intra): n=b*512+t, q in [0,65)  -> out[b][c][t][q]
// mode 1 (inter): n=b*65+qf, q=tt in [0,512) -> out[b][c][tt][qf]
// ---------------------------------------------------------------------------
__global__ __launch_bounds__(256) void k_convT_res(
    const _Float16* __restrict__ hs, const _Float16* __restrict__ lwT,
    const float* __restrict__ lb, const float* __restrict__ resid,
    float* __restrict__ out, int L, int Qout, int mode) {
  const int n = blockIdx.x;
  for (int idx = threadIdx.x; idx < Cc * Qout; idx += 256) {
    int c = idx / Qout, q = idx % Qout;
    float acc = lb[c];
#pragma unroll
    for (int k = 0; k < 4; ++k) {
      int l = q - k;
      if (l < 0 || l >= L) continue;
      const _Float16* hp = hs + (((size_t)n) * L + l) * (2 * Hh);
      const _Float16* wp = lwT + (((size_t)c) * 4 + k) * (2 * Hh);
      __builtin_prefetch(hp + 192, 0, 1);  // -> global_prefetch_b8
      float a = 0.f;
      for (int j = 0; j < 2 * Hh; ++j) a += (float)wp[j] * (float)hp[j];
      acc += a;
    }
    size_t o;
    if (mode == 0) {
      int b = n >> 9, t = n & 511;
      o = (((size_t)b * Cc + c) * Tt + t) * Ffq + q;
    } else {
      int b = n / Ffq, qf = n % Ffq;
      o = (((size_t)b * Cc + c) * Tt + q) * Ffq + qf;
    }
    out[o] = acc + resid[o];
  }
}

// ---------------------------------------------------------------------------
// Per-head QKV projection + PReLU + LN over (e,f); emits f16 GEMM operands:
// q2/k2 [hb][t][EFP] (zero-padded K), v2T [hb][DFP][t] (transposed, padded N)
// ---------------------------------------------------------------------------
__global__ __launch_bounds__(256) void k_qkv(
    const float* __restrict__ inter,
    const float* __restrict__ qw, const float* __restrict__ qb,
    const float* __restrict__ qa, const float* __restrict__ qg,
    const float* __restrict__ qbe,
    const float* __restrict__ kw, const float* __restrict__ kb,
    const float* __restrict__ ka, const float* __restrict__ kg,
    const float* __restrict__ kbe,
    const float* __restrict__ vw, const float* __restrict__ vb,
    const float* __restrict__ va, const float* __restrict__ vg,
    const float* __restrict__ vbe,
    _Float16* __restrict__ q2, _Float16* __restrict__ k2,
    _Float16* __restrict__ v2T) {
  __shared__ float sx[Cc][Ffq];
  __shared__ float yb_[Cc][Ffq];
  __shared__ float red[256];
  const int tid = threadIdx.x;
  const int b = blockIdx.x >> 9, t = blockIdx.x & 511;
  for (int i = tid; i < Cc * Ffq; i += 256) {
    int c = i / Ffq, f = i % Ffq;
    sx[c][f] = inter[(((size_t)b * Cc + c) * Tt + t) * Ffq + f];
  }
  __syncthreads();
  const float* Ws[2] = {qw, kw};
  const float* Bs[2] = {qb, kb};
  const float* As[2] = {qa, ka};
  const float* Gs[2] = {qg, kg};
  const float* Esb[2] = {qbe, kbe};
  _Float16* Os[2] = {q2, k2};
  for (int qk = 0; qk < 2; ++qk) {
    for (int i = tid; i < NHh * Ee * Ffq; i += 256) {
      int h = i / (Ee * Ffq), rem = i % (Ee * Ffq);
      int e = rem / Ffq, f = rem % Ffq;
      float acc = Bs[qk][h * Ee + e];
      for (int c = 0; c < Cc; ++c) acc += Ws[qk][(h * Ee + e) * Cc + c] * sx[c][f];
      float a = As[qk][h];
      yb_[h * Ee + e][f] = acc >= 0.f ? acc : a * acc;
    }
    __syncthreads();
    for (int h = 0; h < NHh; ++h) {
      float s = 0.f, ss = 0.f;
      for (int i = tid; i < EF; i += 256) {
        float v = yb_[h * Ee + i / Ffq][i % Ffq];
        s += v; ss += v * v;
      }
      s = blk_sum(s, red);
      ss = blk_sum(ss, red);
      float mu = s / (float)EF;
      float rs = rsqrtf(ss / (float)EF - mu * mu + EPSV);
      _Float16* outp = Os[qk] + (((size_t)(h * Bn + b)) * Tt + t) * EFP;
      for (int i = tid; i < EF; i += 256) {
        int e = i / Ffq, f = i % Ffq;
        float v = (yb_[h * Ee + e][f] - mu) * rs * Gs[qk][(h * Ee + e) * Ffq + f] +
                  Esb[qk][(h * Ee + e) * Ffq + f];
        outp[i] = (_Float16)v;
      }
      if (tid < EFP - EF) outp[EF + tid] = (_Float16)0.f;
      __syncthreads();
    }
  }
  // V
  for (int i = tid; i < NHh * Dd * Ffq; i += 256) {
    int h = i / (Dd * Ffq), rem = i % (Dd * Ffq);
    int d = rem / Ffq, f = rem % Ffq;
    float acc = vb[h * Dd + d];
    for (int c = 0; c < Cc; ++c) acc += vw[(h * Dd + d) * Cc + c] * sx[c][f];
    float a = va[h];
    yb_[h * Dd + d][f] = acc >= 0.f ? acc : a * acc;
  }
  __syncthreads();
  for (int h = 0; h < NHh; ++h) {
    float s = 0.f, ss = 0.f;
    for (int i = tid; i < DF; i += 256) {
      float v = yb_[h * Dd + i / Ffq][i % Ffq];
      s += v; ss += v * v;
    }
    s = blk_sum(s, red);
    ss = blk_sum(ss, red);
    float mu = s / (float)DF;
    float rs = rsqrtf(ss / (float)DF - mu * mu + EPSV);
    _Float16* outp = v2T + ((size_t)(h * Bn + b)) * DFP * Tt;
    for (int i = tid; i < DF; i += 256) {
      int d = i / Ffq, f = i % Ffq;
      float v = (yb_[h * Dd + d][f] - mu) * rs * vg[(h * Dd + d) * Ffq + f] +
                vbe[(h * Dd + d) * Ffq + f];
      outp[(size_t)i * Tt + t] = (_Float16)v;
    }
    if (tid < DFP - DF) outp[(size_t)(DF + tid) * Tt + t] = (_Float16)0.f;
    __syncthreads();
  }
}

// ---------------------------------------------------------------------------
// Attention scores: S[hb][t][s] = sum_e q2[hb][t][e]*k2[hb][s][e]
// grid (32 Mtiles, 4, 8 hb), 8 waves/block each owning one N tile.
// ---------------------------------------------------------------------------
__global__ __launch_bounds__(256) void k_scores(const _Float16* __restrict__ q2,
                                                const _Float16* __restrict__ k2,
                                                float* __restrict__ S) {
  const int mt = blockIdx.x;
  const int hb = blockIdx.z;
  const int lane = threadIdx.x & 31, wv = threadIdx.x >> 5;
  const int nt = blockIdx.y * 8 + wv;
  const int m = lane & 15, hi = lane >> 4;
  const _Float16* ap = q2 + ((size_t)hb * Tt + mt * 16) * EFP;
  const _Float16* bp = k2 + (size_t)hb * Tt * EFP;
  v8f acc = {};
  for (int kt = 0; kt < EFP / 32; ++kt)
    acc = wmma16(frag_a(ap, EFP, m, hi, kt), frag_b(bp, EFP, nt * 16 + m, hi, kt), acc);
  float* Sp = S + ((size_t)hb * Tt + mt * 16) * Tt + nt * 16;
#pragma unroll
  for (int r = 0; r < 8; ++r) Sp[(size_t)(r + 8 * hi) * Tt + m] = acc[r];
}

// Row softmax with scale 1/sqrt(E*F); emits f16 probabilities for AV WMMA.
__global__ __launch_bounds__(256) void k_softmax(const float* __restrict__ S,
                                                 _Float16* __restrict__ P) {
  __shared__ float red[256];
  const int tid = threadIdx.x;
  const size_t r = blockIdx.x;  // hb*512 + t
  const float* row = S + r * (size_t)Tt;
  const float scale = rsqrtf((float)EF);
  float v0 = row[tid] * scale, v1 = row[tid + 256] * scale;
  red[tid] = fmaxf(v0, v1);
  __syncthreads();
  for (int s = 128; s > 0; s >>= 1) {
    if (tid < s) red[tid] = fmaxf(red[tid], red[tid + s]);
    __syncthreads();
  }
  float mx = red[0];
  __syncthreads();
  float e0 = __expf(v0 - mx), e1 = __expf(v1 - mx);
  float inv = 1.0f / blk_sum(e0 + e1, red);
  P[r * Tt + tid] = (_Float16)(e0 * inv);
  P[r * Tt + tid + 256] = (_Float16)(e1 * inv);
}

// O[hb][t][n] = sum_s P[hb][t][s]*v2T[hb][n][s] ; N tiles = 49 (DFP/16)
__global__ __launch_bounds__(256) void k_av(const _Float16* __restrict__ P,
                                            const _Float16* __restrict__ v2T,
                                            float* __restrict__ O) {
  const int mt = blockIdx.x;
  const int hb = blockIdx.z;
  const int lane = threadIdx.x & 31, wv = threadIdx.x >> 5;
  const int nt = blockIdx.y * 8 + wv;
  if (nt >= DFP / 16) return;  // wave-uniform; EXEC stays all-ones per wave
  const int m = lane & 15, hi = lane >> 4;
  const _Float16* ap = P + ((size_t)hb * Tt + mt * 16) * Tt;
  const _Float16* bp = v2T + (size_t)hb * DFP * Tt;
  v8f acc = {};
  for (int kt = 0; kt < Tt / 32; ++kt)
    acc = wmma16(frag_a(ap, Tt, m, hi, kt), frag_b(bp, Tt, nt * 16 + m, hi, kt), acc);
  float* Op = O + ((size_t)hb * Tt + mt * 16) * DFP + nt * 16;
#pragma unroll
  for (int r = 0; r < 8; ++r) Op[(size_t)(r + 8 * hi) * DFP + m] = acc[r];
}

// ---------------------------------------------------------------------------
// Output projection + PReLU + LN over (c,f) + residual -> d_out
// ---------------------------------------------------------------------------
__global__ __launch_bounds__(256) void k_proj(
    const float* __restrict__ O, const float* __restrict__ inter,
    const float* __restrict__ pw, const float* __restrict__ pb,
    const float* __restrict__ pa, const float* __restrict__ pg,
    const float* __restrict__ pbe, float* __restrict__ out) {
  __shared__ float so[Cc][Ffq];
  __shared__ float sp[Cc][Ffq];
  __shared__ float red[256];
  const int tid = threadIdx.x;
  const int b = blockIdx.x >> 9, t = blockIdx.x & 511;
  for (int i = tid; i < Cc * Ffq; i += 256) {
    int c = i / Ffq, f = i % Ffq;
    int h = c / Dd, d = c % Dd;
    so[c][f] = O[(((size_t)(h * Bn + b)) * Tt + t) * DFP + d * Ffq + f];
  }
  __syncthreads();
  const float a = pa[0];
  for (int i = tid; i < Cc * Ffq; i += 256) {
    int co = i / Ffq, f = i % Ffq;
    float acc = pb[co];
    for (int c = 0; c < Cc; ++c) acc += pw[co * Cc + c] * so[c][f];
    sp[co][f] = acc >= 0.f ? acc : a * acc;
  }
  __syncthreads();
  float s = 0.f, ss = 0.f;
  for (int i = tid; i < Cc * Ffq; i += 256) {
    float v = sp[i / Ffq][i % Ffq];
    s += v; ss += v * v;
  }
  s = blk_sum(s, red);
  ss = blk_sum(ss, red);
  const float mu = s / (float)(Cc * Ffq);
  const float rs = rsqrtf(ss / (float)(Cc * Ffq) - mu * mu + EPSV);
  for (int i = tid; i < Cc * Ffq; i += 256) {
    int co = i / Ffq, f = i % Ffq;
    size_t o = (((size_t)b * Cc + co) * Tt + t) * Ffq + f;
    out[o] = (sp[co][f] - mu) * rs * pg[co * Ffq + f] + pbe[co * Ffq + f] + inter[o];
  }
}

// ---------------------------------------------------------------------------
// Host orchestration
// ---------------------------------------------------------------------------
extern "C" void kernel_launch(void* const* d_in, const int* in_sizes, int n_in,
                              void* d_out, int out_size, void* d_ws, size_t ws_size,
                              hipStream_t stream) {
  const float* x         = (const float*)d_in[0];
  const float* intra_g   = (const float*)d_in[2];
  const float* intra_b   = (const float*)d_in[3];
  const float* intra_wih = (const float*)d_in[4];
  const float* intra_whh = (const float*)d_in[5];
  const float* intra_bih = (const float*)d_in[6];
  const float* intra_bhh = (const float*)d_in[7];
  const float* intra_lw  = (const float*)d_in[8];
  const float* intra_lb  = (const float*)d_in[9];
  const float* inter_g   = (const float*)d_in[10];
  const float* inter_b   = (const float*)d_in[11];
  const float* inter_wih = (const float*)d_in[12];
  const float* inter_whh = (const float*)d_in[13];
  const float* inter_bih = (const float*)d_in[14];
  const float* inter_bhh = (const float*)d_in[15];
  const float* inter_lw  = (const float*)d_in[16];
  const float* inter_lb  = (const float*)d_in[17];
  const float* qw = (const float*)d_in[18];
  const float* qb = (const float*)d_in[19];
  const float* qa = (const float*)d_in[20];
  const float* qg = (const float*)d_in[21];
  const float* qbe = (const float*)d_in[22];
  const float* kw = (const float*)d_in[23];
  const float* kb = (const float*)d_in[24];
  const float* ka = (const float*)d_in[25];
  const float* kg = (const float*)d_in[26];
  const float* kbe = (const float*)d_in[27];
  const float* vw = (const float*)d_in[28];
  const float* vb = (const float*)d_in[29];
  const float* va = (const float*)d_in[30];
  const float* vg = (const float*)d_in[31];
  const float* vbe = (const float*)d_in[32];
  const float* pw = (const float*)d_in[33];
  const float* pb = (const float*)d_in[34];
  const float* pa = (const float*)d_in[35];
  const float* pg = (const float*)d_in[36];
  const float* pbe = (const float*)d_in[37];
  float* out = (float*)d_out;

  char* ws = (char*)d_ws;
  size_t off = 0;
  auto alloc = [&](size_t bytes) -> void* {
    void* p = ws + off;
    off = (off + bytes + 255) & ~((size_t)255);
    return p;
  };

  const size_t MROWS_A = (size_t)NA * LA;    // 63488
  const size_t MROWS_B = (size_t)NBP * LB;   // 73296

  _Float16* wi16A = (_Float16*)alloc((size_t)2 * G4 * CIN * 2);
  _Float16* wh16A = (_Float16*)alloc((size_t)2 * G4 * Hh * 2);
  _Float16* wi16B = (_Float16*)alloc((size_t)2 * G4 * CIN * 2);
  _Float16* wh16B = (_Float16*)alloc((size_t)2 * G4 * Hh * 2);
  _Float16* lwTA  = (_Float16*)alloc((size_t)48 * 4 * 384 * 2);
  _Float16* lwTB  = (_Float16*)alloc((size_t)48 * 4 * 384 * 2);
  float* bsumA = (float*)alloc((size_t)2 * G4 * 4);
  float* bsumB = (float*)alloc((size_t)2 * G4 * 4);
  // xs / hs / xg shared between the two sequential BiLSTM phases
  _Float16* xsS = (_Float16*)alloc(MROWS_B * CIN * 2);        // >= MROWS_A*CIN
  _Float16* hsS = (_Float16*)alloc(MROWS_B * 2 * Hh * 2);     // >= intra size
  _Float16* xgS = (_Float16*)alloc((size_t)2 * MROWS_B * G4 * 2);
  float* intra = (float*)alloc((size_t)Bn * Cc * Tt * Ffq * 4);
  float* inter = (float*)alloc((size_t)Bn * Cc * Tt * Ffq * 4);
  _Float16* q2 = (_Float16*)alloc((size_t)8 * Tt * EFP * 2);
  _Float16* k2 = (_Float16*)alloc((size_t)8 * Tt * EFP * 2);
  _Float16* v2T = (_Float16*)alloc((size_t)8 * DFP * Tt * 2);
  float* S = (float*)alloc((size_t)8 * Tt * Tt * 4);
  _Float16* P = (_Float16*)alloc((size_t)8 * Tt * Tt * 2);
  float* O = (float*)alloc((size_t)8 * Tt * DFP * 4);

  const int nw = 2 * G4 * CIN;  // 294912
  k_f2h<<<(nw + 255) / 256, 256, 0, stream>>>(intra_wih, wi16A, nw);
  k_f2h<<<(nw + 255) / 256, 256, 0, stream>>>(intra_whh, wh16A, nw);
  k_f2h<<<(nw + 255) / 256, 256, 0, stream>>>(inter_wih, wi16B, nw);
  k_f2h<<<(nw + 255) / 256, 256, 0, stream>>>(inter_whh, wh16B, nw);
  k_lwT<<<(384 * 48 * 4 + 255) / 256, 256, 0, stream>>>(intra_lw, lwTA);
  k_lwT<<<(384 * 48 * 4 + 255) / 256, 256, 0, stream>>>(inter_lw, lwTB);
  k_bsum<<<(2 * G4 + 255) / 256, 256, 0, stream>>>(intra_bih, intra_bhh, bsumA, 2 * G4);
  k_bsum<<<(2 * G4 + 255) / 256, 256, 0, stream>>>(inter_bih, inter_bhh, bsumB, 2 * G4);

  // ---- intra path (BiLSTM over F) ----
  k_ln_unfold_intra<<<NA, 256, 0, stream>>>(x, intra_g, intra_b, xsS);
  k_xgemm<<<dim3((unsigned)(MROWS_A / 16), 6, 2), 256, 0, stream>>>(
      xsS, wi16A, bsumA, xgS, (int)MROWS_A);
  k_lstm_rec<<<dim3(NA / 16, 2), 256, LSTM_LDS_TOT, stream>>>(
      xgS, wh16A, hsS, NA, LA);
  k_convT_res<<<NA, 256, 0, stream>>>(hsS, lwTA, intra_lb, x, intra, LA, Ffq, 0);

  // ---- inter path (BiLSTM over T) ----
  k_ln_unfold_inter<<<NBP, 256, 0, stream>>>(intra, inter_g, inter_b, xsS);
  k_xgemm<<<dim3((unsigned)(MROWS_B / 16), 6, 2), 256, 0, stream>>>(
      xsS, wi16B, bsumB, xgS, (int)MROWS_B);
  k_lstm_rec<<<dim3(NBP / 16, 2), 256, LSTM_LDS_TOT, stream>>>(
      xgS, wh16B, hsS, NBP, LB);
  k_convT_res<<<NB, 256, 0, stream>>>(hsS, lwTB, inter_lb, intra, inter, LB, Tt, 1);

  // ---- attention ----
  k_qkv<<<NA, 256, 0, stream>>>(inter, qw, qb, qa, qg, qbe, kw, kb, ka, kg, kbe,
                                vw, vb, va, vg, vbe, q2, k2, v2T);
  k_scores<<<dim3(Tt / 16, 4, 8), 256, 0, stream>>>(q2, k2, S);
  k_softmax<<<8 * Tt, 256, 0, stream>>>(S, P);
  k_av<<<dim3(Tt / 16, 7, 8), 256, 0, stream>>>(P, v2T, O);

  // ---- output projection ----
  k_proj<<<NA, 256, 0, stream>>>(O, inter, pw, pb, pa, pg, pbe, out);
}
